// GGNNRel_35862976921808
// MI455X (gfx1250) — matrix operand order repaired and checked by reading, not compiled, not run on hardware
//
#include <hip/hip_runtime.h>
#include <hip/hip_bf16.h>

typedef __attribute__((ext_vector_type(16))) __bf16 bf16x16;
typedef __attribute__((ext_vector_type(8)))  float  f32x8;

#define R_DIM   2048
#define NODE_D  53
#define H_DIM   512
#define C_DIM   51
#define NOBJ_D  151
#define OUT_D   512
#define M_ROWS  (R_DIM * NODE_D)        // 108544, divisible by 128
#define KOUT    (NODE_D * OUT_D)        // 27136

#define BM 128
#define BN 128
#define BK 64
#define LDK (BK + 8)                    // padded LDS stride (bf16 elems)

union FragAB { bf16x16 v; uint4 q[2]; };

__device__ __forceinline__ unsigned lds_off(const void* p) {
    // generic LDS address = {SHARED_BASE[63:32], lds_byte_offset[31:0]}
    return (unsigned)(unsigned long long)p;
}

// ---------------------------------------------------------------------------
// Tiled bf16 WMMA GEMM:  C[M,N] = A[M,K] * B[N,K]^T  with fused epilogues.
// Double-buffered LDS; staging via async global->LDS (ASYNCcnt) so global
// traffic overlaps WMMA issue and never transits VGPRs.
// A is split: k < kSplit reads A0 (ld=kSplit), else A1 (ld=K-kSplit).
// MODE 0: GRU phase A  (N=1536): zv=sig, rvfh=sig*fh, h-preact
// MODE 1: GRU phase B  (N=512):  hv=tanh(acc+ha); h = (1-zv)h + zv hv (in place)
// MODE 2: output head  (N=512):  relu(acc + bias)
// ---------------------------------------------------------------------------
template<int MODE>
__global__ __launch_bounds__(256)
void gemm_bf16(const __bf16* A0, const __bf16* A1, int kSplit, int K,
               const __bf16* __restrict__ B, const float* __restrict__ bias,
               const __bf16* habf,   // MODE1: read preact
               __bf16* hawr,         // MODE0: write preact
               __bf16* zvbf,         // MODE0 write / MODE1 read
               __bf16* rvfh,         // MODE0 write
               __bf16* hbf,          // MODE0 read fh / MODE1 read+write state
               __bf16* outbf)        // MODE2 write
{
    __shared__ __align__(16) __bf16 Als[2][BM][LDK];
    __shared__ __align__(16) __bf16 Bls[2][BN][LDK];

    const int tid   = threadIdx.x;
    const int lane  = tid & 31;
    const int wave  = tid >> 5;
    const int waveM = wave & 1;          // 2 waves over M (64 rows each)
    const int waveN = wave >> 1;         // 4 waves over N (32 cols each)
    const int rowBase = blockIdx.y * BM;
    const int colBase = blockIdx.x * BN;
    const int ldA1 = K - kSplit;

    f32x8 acc[4][2];
    const f32x8 zf = {0.f,0.f,0.f,0.f,0.f,0.f,0.f,0.f};
#pragma unroll
    for (int i = 0; i < 4; ++i)
#pragma unroll
        for (int j = 0; j < 2; ++j) acc[i][j] = zf;

    // Per-thread staging coordinates: 1024 16B chunks per tile, 4 per thread.
    // chunk -> (row = chunk/8, kc = (chunk%8)*8); same mapping for A and B.
    auto stage = [&](int buf, int k0) {
#pragma unroll
        for (int i = 0; i < 4; ++i) {
            int chunk = tid + i * 256;
            int m  = chunk >> 3;
            int kc = (chunk & 7) << 3;
            int gk = k0 + kc;
            const __bf16* srcA = (gk < kSplit)
                ? (A0 + (size_t)(rowBase + m) * kSplit + gk)
                : (A1 + (size_t)(rowBase + m) * ldA1 + (gk - kSplit));
            asm volatile("global_load_async_to_lds_b128 %0, %1, off"
                         :: "v"(lds_off(&Als[buf][m][kc])),
                            "v"((unsigned long long)srcA)
                         : "memory");
            const __bf16* srcB = B + (size_t)(colBase + m) * K + gk;
            asm volatile("global_load_async_to_lds_b128 %0, %1, off"
                         :: "v"(lds_off(&Bls[buf][m][kc])),
                            "v"((unsigned long long)srcB)
                         : "memory");
        }
    };

    stage(0, 0);
    asm volatile("s_wait_asynccnt 0x0" ::: "memory");
    __syncthreads();

    const int nsteps = K / BK;
    for (int s = 0; s < nsteps; ++s) {
        const int buf = s & 1;
        if (s + 1 < nsteps) stage(buf ^ 1, (s + 1) * BK);   // overlap next tile

#pragma unroll
        for (int ks = 0; ks < BK / 32; ++ks) {
            // pre-load ALL fragments for this k-step, then batch the WMMAs
            FragAB bf[2], af[4];
#pragma unroll
            for (int j = 0; j < 2; ++j) {
                int n  = waveN * 32 + j * 16 + (lane & 15);
                int kk = ks * 32 + (lane >> 4) * 16;
                bf[j].q[0] = *(const uint4*)&Bls[buf][n][kk];
                bf[j].q[1] = *(const uint4*)&Bls[buf][n][kk + 8];
            }
#pragma unroll
            for (int i = 0; i < 4; ++i) {
                int m  = waveM * 64 + i * 16 + (lane & 15);
                int kh = (lane >> 4) * 16;
                af[i].q[0] = *(const uint4*)&Als[buf][m][ks * 32 + kh];
                af[i].q[1] = *(const uint4*)&Als[buf][m][ks * 32 + kh + 8];
            }
#pragma unroll
            for (int i = 0; i < 4; ++i)
#pragma unroll
                for (int j = 0; j < 2; ++j)
                    acc[i][j] = __builtin_amdgcn_wmma_f32_16x16x32_bf16(
                        false, af[i].v, false, bf[j].v, (short)0, acc[i][j],
                        false, false);
        }

        asm volatile("s_wait_asynccnt 0x0" ::: "memory");
        __syncthreads();
    }

    // ---- epilogue. 16x16 f32 C layout: lane<16 -> M=v, lane>=16 -> M=8+v, N=lane&15
#pragma unroll
    for (int i = 0; i < 4; ++i)
#pragma unroll
        for (int j = 0; j < 2; ++j) {
            int mb = rowBase + waveM * 64 + i * 16 + 8 * (lane >> 4);
            int n  = colBase + waveN * 32 + j * 16 + (lane & 15);
#pragma unroll
            for (int v = 0; v < 8; ++v) {
                int m = mb + v;
                float x = acc[i][j][v];
                if constexpr (MODE == 0) {
                    x += bias[n];
                    if (n < 512) {
                        zvbf[(size_t)m * 512 + n] =
                            (__bf16)(1.f / (1.f + __expf(-x)));
                    } else if (n < 1024) {
                        int n2 = n - 512;
                        float rv = 1.f / (1.f + __expf(-x));
                        float fh = (float)hbf[(size_t)m * 512 + n2];
                        rvfh[(size_t)m * 512 + n2] = (__bf16)(rv * fh);
                    } else {
                        hawr[(size_t)m * 512 + (n - 1024)] = (__bf16)x;
                    }
                } else if constexpr (MODE == 1) {
                    size_t idx = (size_t)m * 512 + n;
                    float hv = tanhf(x + (float)habf[idx]);
                    float zv = (float)zvbf[idx];
                    float fh = (float)hbf[idx];
                    hbf[idx] = (__bf16)((1.f - zv) * fh + zv * hv);
                } else {
                    float y = x + bias[n];
                    outbf[(size_t)m * 512 + n] = (__bf16)(y > 0.f ? y : 0.f);
                }
            }
        }
}

// ---------------------------------------------------------------------------
// av construction:  av[0..1] = row . hidden[2:],  av[2+c] = row[c]*(h0+h1)
// ---------------------------------------------------------------------------
__global__ __launch_bounds__(256)
void compute_av(const float* __restrict__ rowf, const __bf16* __restrict__ hbf,
                __bf16* __restrict__ avbf)
{
    __shared__ float rsh[C_DIM];
    int r = blockIdx.x, tid = threadIdx.x;
    if (tid < C_DIM) rsh[tid] = rowf[r * C_DIM + tid];
    __syncthreads();

    const __bf16* hrow = hbf + (size_t)r * NODE_D * H_DIM;
    __bf16* arow = avbf + (size_t)r * NODE_D * H_DIM;
    int h0 = tid * 2;

    float s0 = (float)hrow[h0]     + (float)hrow[H_DIM + h0];
    float s1 = (float)hrow[h0 + 1] + (float)hrow[H_DIM + h0 + 1];
    float a0 = 0.f, a1 = 0.f;
#pragma unroll 3
    for (int c = 0; c < C_DIM; ++c) {
        float w = rsh[c];
        a0 += w * (float)hrow[(size_t)(2 + c) * H_DIM + h0];
        a1 += w * (float)hrow[(size_t)(2 + c) * H_DIM + h0 + 1];
    }
    arow[h0]             = (__bf16)a0;  arow[h0 + 1]             = (__bf16)a1;
    arow[H_DIM + h0]     = (__bf16)a0;  arow[H_DIM + h0 + 1]     = (__bf16)a1;
    for (int c = 0; c < C_DIM; ++c) {
        float w = rsh[c];
        arow[(size_t)(2 + c) * H_DIM + h0]     = (__bf16)(w * s0);
        arow[(size_t)(2 + c) * H_DIM + h0 + 1] = (__bf16)(w * s1);
    }
}

// ---------------------------------------------------------------------------
__global__ void gather_rows(const int* __restrict__ preds,
                            const float* __restrict__ matrix,
                            float* __restrict__ rowf)
{
    int r = blockIdx.x, c = threadIdx.x;
    if (c < C_DIM) {
        int s = preds[2 * r], o = preds[2 * r + 1];
        rowf[r * C_DIM + c] = matrix[((size_t)s * NOBJ_D + o) * C_DIM + c];
    }
}

__global__ void f2bf(const float* __restrict__ src, __bf16* __restrict__ dst,
                     size_t n)
{
    size_t i = (size_t)blockIdx.x * blockDim.x + threadIdx.x;
    size_t st = (size_t)gridDim.x * blockDim.x;
    for (; i < n; i += st) dst[i] = (__bf16)src[i];
}

// Build stacked weight [W3eff|W3u ; W4eff|W4u ; W5eff|0] (1536x1024) + biases.
__global__ void build_bstack(const float* w3w, const float* w3u,
                             const float* w4w, const float* w4u,
                             const float* w5w,
                             const float* b3w, const float* b3u,
                             const float* b4w, const float* b4u,
                             const float* b5w, const float* b5u,
                             __bf16* Bstack, float* biasS)
{
    int idx = blockIdx.x * blockDim.x + threadIdx.x;
    if (idx >= 1536 * 1024) return;
    int r = idx >> 10, k = idx & 1023;
    int g = r >> 9, n = r & 511;
    float v;
    if (g == 0)      v = (k < 512) ? (w3w[n*1024+k] + w3w[n*1024+k+512]) : w3u[n*512 + (k-512)];
    else if (g == 1) v = (k < 512) ? (w4w[n*1024+k] + w4w[n*1024+k+512]) : w4u[n*512 + (k-512)];
    else             v = (k < 512) ? (w5w[n*1024+k] + w5w[n*1024+k+512]) : 0.f;
    Bstack[idx] = (__bf16)v;
    if (k == 0)
        biasS[r] = (g == 0) ? (b3w[n] + b3u[n])
                 : (g == 1) ? (b4w[n] + b4u[n])
                            : (b5w[n] + b5u[n]);
}

// Final projection: rel_dists[r, c] = bc[c] + out_row(27136) . wc[c]
__global__ __launch_bounds__(256)
void final_proj(const __bf16* __restrict__ outbf, const __bf16* __restrict__ wcbf,
                const float* __restrict__ bc, float* __restrict__ dout)
{
    __shared__ __bf16 s[KOUT];
    int r = blockIdx.x;
    const __bf16* src = outbf + (size_t)r * KOUT;
    for (int i = threadIdx.x; i < KOUT / 2; i += 256)
        ((unsigned int*)s)[i] = ((const unsigned int*)src)[i];
    __syncthreads();

    int lane = threadIdx.x & 31, wave = threadIdx.x >> 5;
    for (int c = wave; c < C_DIM; c += 8) {
        const __bf16* w = wcbf + (size_t)c * KOUT;
        float local = 0.f;
        for (int k = lane; k < KOUT; k += 32)
            local += (float)s[k] * (float)w[k];
        for (int off = 16; off > 0; off >>= 1)
            local += __shfl_down(local, off, 32);
        if (lane == 0) dout[(size_t)r * C_DIM + c] = local + bc[c];
    }
}

// ---------------------------------------------------------------------------
extern "C" void kernel_launch(void* const* d_in, const int* in_sizes, int n_in,
                              void* d_out, int out_size, void* d_ws, size_t ws_size,
                              hipStream_t stream)
{
    const int*   preds  = (const int*)  d_in[1];
    const float* input  = (const float*)d_in[2];
    const float* matrix = (const float*)d_in[4];
    const float* w3w = (const float*)d_in[5],  *b3w = (const float*)d_in[6];
    const float* w3u = (const float*)d_in[7],  *b3u = (const float*)d_in[8];
    const float* w4w = (const float*)d_in[9],  *b4w = (const float*)d_in[10];
    const float* w4u = (const float*)d_in[11], *b4u = (const float*)d_in[12];
    const float* w5w = (const float*)d_in[13], *b5w = (const float*)d_in[14];
    const float* w5u = (const float*)d_in[15], *b5u = (const float*)d_in[16];
    const float* wo  = (const float*)d_in[17], *bo  = (const float*)d_in[18];
    const float* wc  = (const float*)d_in[19], *bc  = (const float*)d_in[20];
    float* dout = (float*)d_out;

    const size_t NE = (size_t)M_ROWS * H_DIM;   // 55,574,528 elements
    char* ws = (char*)d_ws;
    size_t off = 0;
    auto alloc = [&](size_t bytes) -> void* {
        off = (off + 255) & ~(size_t)255;
        void* p = ws + off;
        off += bytes;
        return p;
    };
    __bf16* hbf    = (__bf16*)alloc(NE * 2);            // hidden state (bf16)
    __bf16* avbf   = (__bf16*)alloc(NE * 2);            // av_half; later reused as input-bf16
    __bf16* zvbf   = (__bf16*)alloc(NE * 2);
    __bf16* rvfh   = (__bf16*)alloc(NE * 2);            // later reused as relu-out
    __bf16* habf   = (__bf16*)alloc(NE * 2);
    float*  rowf   = (float*) alloc((size_t)R_DIM * C_DIM * 4);
    __bf16* Bstack = (__bf16*)alloc((size_t)1536 * 1024 * 2);
    float*  biasS  = (float*) alloc(1536 * 4);
    __bf16* w5ubf  = (__bf16*)alloc((size_t)512 * 512 * 2);
    __bf16* wobf   = (__bf16*)alloc((size_t)512 * 1024 * 2);
    __bf16* wcbf   = (__bf16*)alloc((size_t)C_DIM * KOUT * 2);
    __bf16* outbf  = rvfh;                              // alias (free after last phase B)
    __bf16* xbf    = avbf;                              // alias (free after last phase A)

    // ---- prep
    f2bf<<<8192, 256, 0, stream>>>(input, hbf, NE);
    build_bstack<<<(1536 * 1024 + 255) / 256, 256, 0, stream>>>(
        w3w, w3u, w4w, w4u, w5w, b3w, b3u, b4w, b4u, b5w, b5u, Bstack, biasS);
    f2bf<<<512, 256, 0, stream>>>(w5u, w5ubf, (size_t)512 * 512);
    f2bf<<<1024, 256, 0, stream>>>(wo, wobf, (size_t)512 * 1024);
    f2bf<<<2048, 256, 0, stream>>>(wc, wcbf, (size_t)C_DIM * KOUT);
    gather_rows<<<R_DIM, 64, 0, stream>>>(preds, matrix, rowf);

    const int gridM = M_ROWS / BM;                      // 848
    // ---- 3 GRU steps
    for (int t = 0; t < 3; ++t) {
        compute_av<<<R_DIM, 256, 0, stream>>>(rowf, hbf, avbf);
        gemm_bf16<0><<<dim3(1536 / BN, gridM), 256, 0, stream>>>(
            avbf, hbf, 512, 1024, Bstack, biasS,
            nullptr, habf, zvbf, rvfh, hbf, nullptr);
        gemm_bf16<1><<<dim3(512 / BN, gridM), 256, 0, stream>>>(
            rvfh, rvfh, 512, 512, w5ubf, biasS,
            habf, nullptr, zvbf, nullptr, hbf, nullptr);
    }

    // ---- output head: relu([hidden | input] @ wo.T + bo)
    f2bf<<<8192, 256, 0, stream>>>(input, xbf, NE);
    gemm_bf16<2><<<dim3(512 / BN, gridM), 256, 0, stream>>>(
        hbf, xbf, 512, 1024, wobf, bo,
        nullptr, nullptr, nullptr, nullptr, nullptr, outbf);

    // ---- rel_dists = out.reshape(R, 27136) @ wc.T + bc
    final_proj<<<R_DIM, 256, 0, stream>>>(outbf, wcbf, bc, dout);
}
